// CausalSelfAttention_3307124818649
// MI455X (gfx1250) — matrix operand order repaired and checked by reading, not compiled, not run on hardware
//
#include <hip/hip_runtime.h>
#include <hip/hip_bf16.h>
#include <stdint.h>

// CDNA5 (gfx1250) causal self-attention pipeline:
//   q/k/v = x @ W^T      : WMMA f16 GEMM (f32 accum), 128x64 tile, 8 WMMA/wave/stage
//   rmsnorm+rope (q,k)   : wave-per-vector, emits f16 (q pre-scaled by 1/sqrt(hd))
//   v transpose          : per-head [t][d] -> [d][t] f16 for TDM-friendly staging
//   flash attention      : Q/K/V tiles staged by tensor_load_to_lds (TDM) with
//                          descriptor padding -> banked LDS strides; QK^T and P@V
//                          via v_wmma_f32_16x16x32_f16; online softmax via width-16
//                          shuffles matching the WMMA C-layout
//   y @ w_proj^T         : same WMMA GEMM, straight into d_out
//   tuple out #2         : v1 passthrough (async D2D)

typedef _Float16 v16h __attribute__((ext_vector_type(16)));
typedef float    v8f  __attribute__((ext_vector_type(8)));
typedef unsigned int u32x4 __attribute__((ext_vector_type(4)));
typedef int          i32x8 __attribute__((ext_vector_type(8)));
typedef int          i32x4 __attribute__((ext_vector_type(4)));

#define WMMA_F16(a, b, c) \
  __builtin_amdgcn_wmma_f32_16x16x32_f16(false, (a), false, (b), (short)0, (c), false, false)

#ifndef __has_builtin
#define __has_builtin(x) 0
#endif
#if defined(__HIP_DEVICE_COMPILE__) && __has_builtin(__builtin_amdgcn_tensor_load_to_lds)
#define HAVE_TDM 1
#else
#define HAVE_TDM 0
#endif

__device__ __forceinline__ _Float16 f2h(float x) { return (_Float16)x; }

__device__ __forceinline__ float rowmax16(float v) {
#pragma unroll
  for (int m = 8; m >= 1; m >>= 1) v = fmaxf(v, __shfl_xor(v, m, 16));
  return v;
}
__device__ __forceinline__ float rowsum16(float v) {
#pragma unroll
  for (int m = 8; m >= 1; m >>= 1) v += __shfl_xor(v, m, 16);
  return v;
}

#if HAVE_TDM
// Issue one 2-D TDM tile load: tile_w contiguous f16 elements per row,
// tile_h rows, row stride stride_elems (f16 units), into LDS at lds_addr.
// LDS padding: (padAmountCode+1) DWORDs inserted every 2^(padIntervalCode+1)
// DWORDs -> reproduces the padded LDS row strides the fragment gathers use.
// 6-arg builtin form: (g0, g1, g2, g3, g4, cpol); 2-D tensor -> g2/g3/g4 zero.
__device__ __forceinline__ void tdm_load_2d(unsigned lds_addr, const void* gptr,
                                            unsigned tile_w, unsigned tile_h,
                                            unsigned stride_elems,
                                            unsigned padIntervalCode,
                                            unsigned padAmountCode) {
  const unsigned long long ga = (unsigned long long)(uintptr_t)gptr;
  const unsigned tensor_w = tile_w;
  const unsigned tensor_h = 1u << 20;          // generous bound; tiles never OOB
  u32x4 g0;
  g0.x = 1u;                                   // count=1, no gather, user mode
  g0.y = lds_addr;                             // lds_addr (bytes)
  g0.z = (unsigned)ga;                         // global_addr[31:0]
  g0.w = (unsigned)(ga >> 32) | (2u << 30);    // global_addr[56:32] | type=2
  i32x8 g1;
  g1[0] = (int)((1u << 16)                     // data_size = 2 bytes
                | (1u << 20)                   // pad_enable
                | (padIntervalCode << 22) | (padAmountCode << 25));
  g1[1] = (int)(tensor_w << 16);               // barrier_addr=0 | tensor_dim0 lo
  g1[2] = (int)((tensor_w >> 16) | ((tensor_h & 0xffffu) << 16));
  g1[3] = (int)((tensor_h >> 16) | (tile_w << 16));   // | tile_dim0
  g1[4] = (int)(tile_h & 0xffffu);             // tile_dim1 | tile_dim2=0
  g1[5] = (int)stride_elems;                   // tensor_dim0_stride lo32
  g1[6] = 0;
  g1[7] = 0;
  i32x4 z4 = {0, 0, 0, 0};
  i32x8 z8 = {0, 0, 0, 0, 0, 0, 0, 0};
  __builtin_amdgcn_tensor_load_to_lds(g0, g1, z4, z4, z8, 0);
}
#endif

// ---------------------------------------------------------------------------
// GEMM: C[M][N] = A[M][K] @ W[N][K]^T (fp32 A/W, f16 WMMA, f32 accum).
// Block tile 128x64, K-chunks of 64. 8 waves; wave w covers rows 32*(w&3),
// cols 32*(w>>2): 2 A-frags x 2 B-frags -> 4 accumulators, 8 WMMAs per stage.
// Output: f32 (Cf) or f16 (Ch); optional epilogue C=(1-lam)*acc+lam*blend.
// ---------------------------------------------------------------------------
#define GT_M 128
#define GT_N 64
#define GT_K 64
#define GLDA 72

__global__ __launch_bounds__(256) void gemm_nt_wmma(
    const float* __restrict__ A, const float* __restrict__ W,
    float* __restrict__ Cf, _Float16* __restrict__ Ch,
    int M, int N, int K,
    const float* __restrict__ blend, const float* __restrict__ lamb) {
  __shared__ _Float16 As[GT_M][GLDA];   // 18 KB
  __shared__ _Float16 Ws[GT_N][GLDA];   //  9 KB

  const int tid  = threadIdx.x;
  const int lane = tid & 31;
  const int w    = tid >> 5;
  const int g    = lane >> 4;
  const int l16  = lane & 15;
  const int wm   = w & 3;
  const int wn   = w >> 2;

  const int m0 = blockIdx.y * GT_M;
  const int n0 = blockIdx.x * GT_N;

  v8f acc[2][2] = {{{}, {}}, {{}, {}}};

  const int arow = tid >> 1, acol = (tid & 1) << 5;   // 32 f32 each
  const int wrow = tid >> 2, wcol = (tid & 3) << 4;   // 16 f32 each

  for (int kc = 0; kc < K; kc += GT_K) {
    const float* aR = A + (size_t)(m0 + arow) * K + kc + acol;
    const float* wR = W + (size_t)(n0 + wrow) * K + kc + wcol;
    if (kc + GT_K < K) {
      __builtin_prefetch(aR + GT_K, 0, 1);
      __builtin_prefetch(wR + GT_K, 0, 1);
    }
#pragma unroll
    for (int j = 0; j < 32; ++j) As[arow][acol + j] = f2h(aR[j]);
#pragma unroll
    for (int j = 0; j < 16; ++j) Ws[wrow][wcol + j] = f2h(wR[j]);
    __syncthreads();

#pragma unroll
    for (int h2 = 0; h2 < 2; ++h2) {
      const int kc2 = h2 << 5;
      v16h a0, a1, b0, b1;
#pragma unroll
      for (int e = 0; e < 16; ++e) {
        const int ka = kc2 + ((e & 8) << 1) + (g << 3) + (e & 7);
        const int kb = kc2 + ((e & 8) << 1) + (g << 3) + (e & 7);
        a0[e] = As[(wm << 5) + l16][ka];
        a1[e] = As[(wm << 5) + 16 + l16][ka];
        b0[e] = Ws[(wn << 5) + l16][kb];
        b1[e] = Ws[(wn << 5) + 16 + l16][kb];
      }
      acc[0][0] = WMMA_F16(a0, b0, acc[0][0]);
      acc[0][1] = WMMA_F16(a0, b1, acc[0][1]);
      acc[1][0] = WMMA_F16(a1, b0, acc[1][0]);
      acc[1][1] = WMMA_F16(a1, b1, acc[1][1]);
    }
    __syncthreads();
  }

  const float lam = blend ? lamb[0] : 0.0f;
#pragma unroll
  for (int i = 0; i < 2; ++i)
#pragma unroll
    for (int j = 0; j < 2; ++j)
#pragma unroll
      for (int r = 0; r < 8; ++r) {
        const int row = m0 + (wm << 5) + (i << 4) + (g << 3) + r;
        const int col = n0 + (wn << 5) + (j << 4) + l16;
        float o = acc[i][j][r];
        if (blend) o = (1.0f - lam) * o + lam * blend[(size_t)row * N + col];
        if (Cf) Cf[(size_t)row * N + col] = o;
        else    Ch[(size_t)row * N + col] = f2h(o);
      }
}

// ---------------------------------------------------------------------------
// RMS-norm (per 64-wide head vector) + RoPE; f32 in, f16 out (with outScale,
// used to fold the attention 1/sqrt(hd) into q). One wave per vector.
// ---------------------------------------------------------------------------
__global__ __launch_bounds__(256) void rmsnorm_rope(
    const float* __restrict__ src, _Float16* __restrict__ dst,
    int T, int H, float outScale) {
  const int lane = threadIdx.x & 31;
  const int wv   = threadIdx.x >> 5;
  const int vi   = blockIdx.x * 8 + wv;        // (b*T + t)*H + h
  const int t    = (vi / H) % T;

  const float* p = src + (size_t)vi * 64;
  const float x1 = p[lane];
  const float x2 = p[lane + 32];

  float ss = x1 * x1 + x2 * x2;
#pragma unroll
  for (int m = 16; m >= 1; m >>= 1) ss += __shfl_xor(ss, m, 32);
  const float rn = rsqrtf(ss * (1.0f / 64.0f) + 1e-6f);

  const float freq = (float)t * __powf(10000.0f, -(float)lane * (1.0f / 32.0f));
  float sn, cs;
  __sincosf(freq, &sn, &cs);

  const float n1 = x1 * rn, n2 = x2 * rn;
  _Float16* q = dst + (size_t)vi * 64;
  q[lane]      = f2h(( n1 * cs + n2 * sn) * outScale);
  q[lane + 32] = f2h((-n1 * sn + n2 * cs) * outScale);
}

// ---------------------------------------------------------------------------
// V transpose: vh[(b*T+t)*D + h*64 + d] -> vt[((b*H+h)*64 + d)*T + t]  (f16).
// 32x32 LDS tiles.
// ---------------------------------------------------------------------------
__global__ __launch_bounds__(256) void transpose_v(
    const _Float16* __restrict__ vh, _Float16* __restrict__ vt, int T, int H) {
  __shared__ _Float16 tile[32][33];
  const int t0 = blockIdx.x * 32;
  const int d0 = blockIdx.y * 32;
  const int bh = blockIdx.z;
  const int b = bh / H, h = bh % H;
  const int D = H * 64;

  const int r  = threadIdx.x >> 3;        // 0..31
  const int c0 = (threadIdx.x & 7) << 2;  // 4 consecutive
#pragma unroll
  for (int j = 0; j < 4; ++j)
    tile[r][c0 + j] = vh[(size_t)(b * T + t0 + r) * D + h * 64 + d0 + c0 + j];
  __syncthreads();
#pragma unroll
  for (int j = 0; j < 4; ++j)
    vt[(size_t)(bh * 64 + d0 + r) * T + t0 + c0 + j] = tile[c0 + j][r];
}

// ---------------------------------------------------------------------------
// Causal flash attention on f16 q/k/vt, f32 output. Block: 128 query rows of
// one (b,h); 8 waves x 16 rows; KV tiles of 32. Tiles staged by TDM (wave 0
// issues, s_wait_tensorcnt, barrier publishes); descriptor padding yields the
// padded LDS strides (72 / 40 halves) assumed by the b128 fragment gathers.
// ---------------------------------------------------------------------------
__global__ __launch_bounds__(256) void flash_attn_wmma(
    const _Float16* __restrict__ Qh, const _Float16* __restrict__ Kh,
    const _Float16* __restrict__ Vt, float* __restrict__ Y, int T, int H) {
  constexpr int QTB = 128, KVT = 32;
  const int Dm = H * 64;

  const int nQB = T / QTB;
  const int qb  = blockIdx.x % nQB;
  const int h   = (blockIdx.x / nQB) % H;
  const int b   = blockIdx.x / (nQB * H);

  __shared__ _Float16 Qs[QTB][72];      // 18 KB (rows of 64 + 8 pad)
  __shared__ _Float16 Ks[KVT][72];      // 4.5 KB
  __shared__ _Float16 Vs[64][40];       // 5 KB, transposed [d][kv], 32 + 8 pad
  __shared__ _Float16 Ps[8][16][40];    // 10 KB, per-wave P tile

  const int tid  = threadIdx.x;
  const int lane = tid & 31;
  const int w    = tid >> 5;
  const int g    = lane >> 4;
  const int l16  = lane & 15;

  const int wS    = __builtin_amdgcn_readfirstlane(w);
  const int qRow0 = qb * QTB + (w << 4);
  const int qHiS  = qb * QTB + (wS << 4) + 15;

  const _Float16* qTile = Qh + (size_t)(b * T + qb * QTB) * Dm + h * 64;

#if HAVE_TDM
  if (wS == 0) {
    tdm_load_2d((unsigned)(uintptr_t)&Qs[0][0], qTile, 64, QTB, Dm, 4, 3);
    __builtin_amdgcn_s_wait_tensorcnt(0);
  }
#else
  {
    const int tr = tid >> 1, tc = (tid & 1) << 5;
#pragma unroll
    for (int j = 0; j < 32; ++j) Qs[tr][tc + j] = qTile[(size_t)tr * Dm + tc + j];
  }
#endif

  v8f o[4] = {{}, {}, {}, {}};
  float mrow[8], lrow[8];
#pragma unroll
  for (int r = 0; r < 8; ++r) { mrow[r] = -1e30f; lrow[r] = 0.0f; }

  const int nTiles = (qb + 1) * (QTB / KVT);
  __syncthreads();

  for (int kt = 0; kt < nTiles; ++kt) {
    const int kv0 = kt * KVT;

#if HAVE_TDM
    if (wS == 0) {
      tdm_load_2d((unsigned)(uintptr_t)&Ks[0][0],
                  Kh + (size_t)(b * T + kv0) * Dm + h * 64, 64, KVT, Dm, 4, 3);
      tdm_load_2d((unsigned)(uintptr_t)&Vs[0][0],
                  Vt + (size_t)(b * H + h) * 64 * T + kv0, KVT, 64, T, 3, 3);
      __builtin_amdgcn_s_wait_tensorcnt(0);
    }
#else
    {
      const int tr = tid >> 3, tc = (tid & 7) << 3;
      const _Float16* kr = Kh + (size_t)(b * T + kv0 + tr) * Dm + h * 64;
#pragma unroll
      for (int j = 0; j < 8; ++j) Ks[tr][tc + j] = kr[tc + j];
      const int vr = tid >> 2, vc = (tid & 3) << 3;
      const _Float16* vp = Vt + (size_t)((b * H + h) * 64 + vr) * T + kv0 + vc;
#pragma unroll
      for (int j = 0; j < 8; ++j) Vs[vr][vc + j] = vp[j];
    }
#endif
    __syncthreads();

    if (kv0 <= qHiS) {                   // wave-uniform: EXEC stays all-ones
      v8f s0 = {}, s1 = {};
#pragma unroll
      for (int c = 0; c < 2; ++c) {
        v16h a, b0, b1;
#pragma unroll
        for (int e = 0; e < 16; ++e) {
          const int ka = ((e & 8) << 1) + (g << 3) + (e & 7);
          const int kb = ((e & 8) << 1) + (g << 3) + (e & 7);
          a[e]  = Qs[(w << 4) + l16][(c << 5) + ka];
          b0[e] = Ks[l16][(c << 5) + kb];
          b1[e] = Ks[16 + l16][(c << 5) + kb];
        }
        s0 = WMMA_F16(a, b0, s0);
        s1 = WMMA_F16(a, b1, s1);
      }

      float alpha[8];
#pragma unroll
      for (int r = 0; r < 8; ++r) {
        const int rowg = qRow0 + r + (g << 3);
        float v0 = s0[r]; if (kv0 + l16 > rowg)      v0 = -1e30f;
        float v1 = s1[r]; if (kv0 + 16 + l16 > rowg) v1 = -1e30f;
        const float mt = rowmax16(fmaxf(v0, v1));
        const float mn = fmaxf(mrow[r], mt);
        const float al = __expf(mrow[r] - mn);
        const float p0 = __expf(v0 - mn);
        const float p1 = __expf(v1 - mn);
        lrow[r] = lrow[r] * al + rowsum16(p0 + p1);
        mrow[r] = mn;
        alpha[r] = al;
        Ps[w][r + (g << 3)][l16]      = f2h(p0);
        Ps[w][r + (g << 3)][16 + l16] = f2h(p1);
      }
#pragma unroll
      for (int r = 0; r < 8; ++r) {
        o[0][r] *= alpha[r]; o[1][r] *= alpha[r];
        o[2][r] *= alpha[r]; o[3][r] *= alpha[r];
      }

      v16h pa;
#pragma unroll
      for (int e = 0; e < 16; ++e)
        pa[e] = Ps[w][l16][((e & 8) << 1) + (g << 3) + (e & 7)];
#pragma unroll
      for (int t4 = 0; t4 < 4; ++t4) {
        v16h bv;
#pragma unroll
        for (int e = 0; e < 16; ++e)
          bv[e] = Vs[(t4 << 4) + l16][((e & 8) << 1) + (g << 3) + (e & 7)];
        o[t4] = WMMA_F16(pa, bv, o[t4]);
      }
    }
    __syncthreads();
  }

  float inv[8];
#pragma unroll
  for (int r = 0; r < 8; ++r) inv[r] = 1.0f / lrow[r];
#pragma unroll
  for (int t4 = 0; t4 < 4; ++t4)
#pragma unroll
    for (int r = 0; r < 8; ++r) {
      const int rw = qRow0 + r + (g << 3);
      Y[(size_t)(b * T + rw) * Dm + h * 64 + (t4 << 4) + l16] = o[t4][r] * inv[r];
    }
}

// ---------------------------------------------------------------------------
// Host launcher. Inputs: x, v1, w_q, w_k, w_v, w_proj, lamb. Out: concat(y, v1).
// ---------------------------------------------------------------------------
extern "C" void kernel_launch(void* const* d_in, const int* in_sizes, int n_in,
                              void* d_out, int out_size, void* d_ws, size_t ws_size,
                              hipStream_t stream) {
  const float* x      = (const float*)d_in[0];
  const float* v1     = (const float*)d_in[1];
  const float* w_q    = (const float*)d_in[2];
  const float* w_k    = (const float*)d_in[3];
  const float* w_v    = (const float*)d_in[4];
  const float* w_proj = (const float*)d_in[5];
  const float* lamb   = (const float*)d_in[6];

  int D = 1;
  while ((long long)D * D < (long long)in_sizes[2]) ++D;   // w_q is D x D
  const int BT = in_sizes[0] / D;
  const int B  = 2;
  const int T  = BT / B;
  const int H  = 16;                                       // hd = 64

  const size_t S = (size_t)BT * D;
  float*    qb = (float*)d_ws;            // f32 q (pre-norm)
  float*    kb = qb + S;                  // f32 k (pre-norm)
  float*    yb = kb + S;                  // f32 attention output
  _Float16* qh = (_Float16*)(yb + S);     // f16 q (normed, roped, scaled)
  _Float16* kh = qh + S;                  // f16 k (normed, roped)
  _Float16* vh = kh + S;                  // f16 v (blended)
  _Float16* vt = vh + S;                  // f16 v transposed per head

  dim3 blk(256);
  dim3 gGemm(D / GT_N, BT / GT_M);

  hipLaunchKernelGGL(gemm_nt_wmma, gGemm, blk, 0, stream, x, w_q, qb,
                     (_Float16*)nullptr, BT, D, D,
                     (const float*)nullptr, (const float*)nullptr);
  hipLaunchKernelGGL(gemm_nt_wmma, gGemm, blk, 0, stream, x, w_k, kb,
                     (_Float16*)nullptr, BT, D, D,
                     (const float*)nullptr, (const float*)nullptr);
  hipLaunchKernelGGL(gemm_nt_wmma, gGemm, blk, 0, stream, x, w_v,
                     (float*)nullptr, vh, BT, D, D, v1, lamb);

  dim3 gNorm((unsigned)((size_t)BT * H / 8));
  hipLaunchKernelGGL(rmsnorm_rope, gNorm, blk, 0, stream, qb, qh, T, H, 0.125f);
  hipLaunchKernelGGL(rmsnorm_rope, gNorm, blk, 0, stream, kb, kh, T, H, 1.0f);

  dim3 gTr(T / 32, 2, B * H);
  hipLaunchKernelGGL(transpose_v, gTr, blk, 0, stream, vh, vt, T, H);

  dim3 gAttn((unsigned)(B * H * (T / 128)));
  hipLaunchKernelGGL(flash_attn_wmma, gAttn, blk, 0, stream, qh, kh, vt, yb, T, H);

  hipLaunchKernelGGL(gemm_nt_wmma, gGemm, blk, 0, stream, yb, w_proj,
                     (float*)d_out, (_Float16*)nullptr, BT, D, D,
                     (const float*)nullptr, (const float*)nullptr);

  (void)hipMemcpyAsync((float*)d_out + S, v1, S * sizeof(float),
                       hipMemcpyDeviceToDevice, stream);
}
